// MambaEncoderLayer_13675175871078
// MI455X (gfx1250) — matrix operand order repaired and checked by reading, not compiled, not run on hardware
//
#include <hip/hip_runtime.h>
#include <cstdint>
#include <cstddef>

typedef __attribute__((ext_vector_type(16))) _Float16 v16h;
typedef __attribute__((ext_vector_type(8)))  float    v8f;
typedef __attribute__((ext_vector_type(4)))  int      v4i;

#define BB 2
#define LL 1024
#define MM (BB*LL)          // 2048 token rows
#define DMODEL 768
#define DINNER 1536
#define DSTATE 16
#define DTRANK 48
#define DTPAD 64
#define NIN (2*DINNER)      // 3072
#define NXP (DTRANK + 2*DSTATE) // 80
#define NCH 8               // scan chunks
#define CH  (LL/NCH)        // 128 steps per chunk

struct alignas(16) U4 { uint32_t a, b, c, d; };

#if defined(__HIP_DEVICE_COMPILE__) && \
    __has_builtin(__builtin_amdgcn_global_load_async_to_lds_b128) && \
    __has_builtin(__builtin_amdgcn_s_wait_asynccnt)
#define GEMM_ASYNC 1
#else
#define GEMM_ASYNC 0
#endif

#if GEMM_ASYNC
__device__ __forceinline__ void async_cp16(const _Float16* g, _Float16* l) {
    __builtin_amdgcn_global_load_async_to_lds_b128(
        (__attribute__((address_space(1))) v4i*)g,
        (__attribute__((address_space(3))) v4i*)l, 0, 0);
}
#endif

// ---------------------------------------------------------------- utilities
__global__ void cvt_rows_kernel(_Float16* __restrict__ dst, int dst_stride,
                                const float* __restrict__ src, int rows, int cols) {
    int i = blockIdx.x * 256 + threadIdx.x;
    if (i >= rows * cols) return;
    int r = i / cols, c = i - r * cols;
    dst[(size_t)r * dst_stride + c] = (_Float16)src[i];
}

__global__ void cvt_pad_kernel(_Float16* __restrict__ dst, int dstw,
                               const float* __restrict__ src, int srcw,
                               int rows, int cols) {
    int i = blockIdx.x * 256 + threadIdx.x;
    if (i >= rows * dstw) return;
    int r = i / dstw, c = i - r * dstw;
    dst[i] = (c < cols) ? (_Float16)src[(size_t)r * srcw + c] : (_Float16)0.0f;
}

__global__ void neg_exp_kernel(float* __restrict__ dst, const float* __restrict__ src, int n) {
    int i = blockIdx.x * 256 + threadIdx.x;
    if (i < n) dst[i] = -__expf(src[i]);
}

// one block per row, 256 threads (8 wave32)
__global__ __launch_bounds__(256) void layernorm_f16_kernel(
    const float* __restrict__ x, const float* __restrict__ w, const float* __restrict__ b,
    _Float16* __restrict__ out, int D) {
    int row = blockIdx.x;
    const float* xr = x + (size_t)row * D;
    float s = 0.f, ss = 0.f;
    for (int c = threadIdx.x; c < D; c += 256) { float v = xr[c]; s += v; ss += v * v; }
    for (int o = 16; o > 0; o >>= 1) { s += __shfl_xor(s, o, 32); ss += __shfl_xor(ss, o, 32); }
    __shared__ float sh_s[8], sh_ss[8];
    int wid = threadIdx.x >> 5, lane = threadIdx.x & 31;
    if (lane == 0) { sh_s[wid] = s; sh_ss[wid] = ss; }
    __syncthreads();
    if (threadIdx.x == 0) {
        float ts = 0.f, tss = 0.f;
        for (int i = 0; i < 8; i++) { ts += sh_s[i]; tss += sh_ss[i]; }
        sh_s[0] = ts; sh_ss[0] = tss;
    }
    __syncthreads();
    float mean = sh_s[0] / D;
    float var  = sh_ss[0] / D - mean * mean;
    float inv  = rsqrtf(var + 1e-5f);
    for (int c = threadIdx.x; c < D; c += 256)
        out[(size_t)row * D + c] = (_Float16)((xr[c] - mean) * inv * w[c] + b[c]);
}

// depthwise causal conv (dir-aware) + bias + SiLU; xz pre-offset per direction
__global__ void conv_silu_kernel(const float* __restrict__ xz,
                                 const float* __restrict__ cw, const float* __restrict__ cb,
                                 float* __restrict__ xc, _Float16* __restrict__ xch, int dir) {
    int i = blockIdx.x * 256 + threadIdx.x;
    if (i >= MM * DINNER) return;
    int d = i % DINNER;
    int t = i / DINNER;                // b*L + l
    int b = t >> 10, l = t & (LL - 1);
    float acc = cb[d];
    if (dir == 0) {
        #pragma unroll
        for (int k = 0; k < 4; k++) {
            int lk = l - 3 + k;
            if (lk >= 0) acc += cw[d * 4 + k] * xz[((size_t)(b * LL + lk)) * NIN + d];
        }
    } else {
        #pragma unroll
        for (int k = 0; k < 4; k++) {
            int lk = l + 3 - k;
            if (lk < LL) acc += cw[d * 4 + k] * xz[((size_t)(b * LL + lk)) * NIN + d];
        }
    }
    float s = acc / (1.f + __expf(-acc));
    xc[i] = s;
    xch[i] = (_Float16)s;
}

// ---------------------------------------------------------------- WMMA GEMM
// C[M,N] = A[M,K] @ W[N,K]^T (+ epilogue), f16 in, f32 out.
// Double-buffered LDS, 64-wide K stage (16 WMMAs/stage, 1 barrier/stage),
// async L2->LDS staging when the toolchain exposes the gfx1250 builtins.
// EPI: 0=none  1=bias+softplus  2=+res  3=bias+GELU(+res)
#define TM 128
#define TN 128
#define TK2 64
#define LDP2 72   // padded LDS row (halves); 144B, 16B aligned, conflict-free

template<int EPI>
__global__ __launch_bounds__(256) void gemm_wmma_kernel(
    const _Float16* __restrict__ A, const _Float16* __restrict__ W,
    float* __restrict__ Cout, const float* __restrict__ bias,
    const float* __restrict__ res, int M, int N, int K) {
    __shared__ alignas(16) _Float16 sm[2][2][TM * LDP2];   // [buf][A|B] = 72KB

    const int tid  = threadIdx.x;
    const int lane = tid & 31;
    const int wave = tid >> 5;          // 8 wave32
    const int wr   = wave >> 2;         // 0..1
    const int wc   = wave & 3;          // 0..3
    const int m0   = blockIdx.x * TM;
    const int n0   = blockIdx.y * TN;

    v8f acc[4][2] = {};

    const int srow = tid >> 1;          // 0..127
    const int scol = (tid & 1) * 32;    // 0 / 32 halves

    union Frag { v16h v; U4 u[2]; };
    const int kha = (lane < 16) ? 0 : 8;
    const int khb = (lane < 16) ? 0 : 8;

#if GEMM_ASYNC
    auto issue = [&](int k0, int buf) {
        const _Float16* ga = A + (size_t)(m0 + srow) * K + k0 + scol;
        _Float16* la = &sm[buf][0][srow * LDP2 + scol];
        #pragma unroll
        for (int j = 0; j < 4; j++) async_cp16(ga + 8 * j, la + 8 * j);
        int n = n0 + srow;
        if (n < N) {
            const _Float16* gb = W + (size_t)n * K + k0 + scol;
            _Float16* lb = &sm[buf][1][srow * LDP2 + scol];
            #pragma unroll
            for (int j = 0; j < 4; j++) async_cp16(gb + 8 * j, lb + 8 * j);
        }
    };
    issue(0, 0);
#else
    U4 rA[4], rB[4];
    auto gload = [&](int k0) {
        const _Float16* ga = A + (size_t)(m0 + srow) * K + k0 + scol;
        #pragma unroll
        for (int j = 0; j < 4; j++) rA[j] = *(const U4*)(ga + 8 * j);
        if (k0 + TK2 < K) __builtin_prefetch(ga + TK2, 0, 1);
        int n = n0 + srow;
        U4 z = {0, 0, 0, 0};
        rB[0] = z; rB[1] = z; rB[2] = z; rB[3] = z;
        if (n < N) {
            const _Float16* gb = W + (size_t)n * K + k0 + scol;
            #pragma unroll
            for (int j = 0; j < 4; j++) rB[j] = *(const U4*)(gb + 8 * j);
            if (k0 + TK2 < K) __builtin_prefetch(gb + TK2, 0, 1);
        }
    };
    gload(0);
#endif

    for (int s = 0, k0 = 0; k0 < K; ++s, k0 += TK2) {
        const int buf = s & 1;
#if GEMM_ASYNC
        __builtin_amdgcn_s_wait_asynccnt(0);   // this wave's stage-s copies landed
        __syncthreads();                       // everyone's landed; prev compute done
        if (k0 + TK2 < K) issue(k0 + TK2, buf ^ 1);   // overlap with WMMA below
#else
        {   // spill staged regs to LDS buffer `buf`
            _Float16* la = &sm[buf][0][srow * LDP2 + scol];
            _Float16* lb = &sm[buf][1][srow * LDP2 + scol];
            #pragma unroll
            for (int j = 0; j < 4; j++) { *(U4*)(la + 8 * j) = rA[j]; *(U4*)(lb + 8 * j) = rB[j]; }
        }
        __syncthreads();
        if (k0 + TK2 < K) gload(k0 + TK2);     // overlap with WMMA below
#endif
        #pragma unroll
        for (int ks = 0; ks < 2; ks++) {
            const int kb = ks * 32;
            Frag a[4], bf[2];
            #pragma unroll
            for (int mt = 0; mt < 4; mt++) {
                const _Float16* rp = &sm[buf][0][(wr * 64 + mt * 16 + (lane & 15)) * LDP2 + kb];
                a[mt].u[0] = *(const U4*)(rp + kha);
                a[mt].u[1] = *(const U4*)(rp + kha + 16);
            }
            #pragma unroll
            for (int nt = 0; nt < 2; nt++) {
                const _Float16* rp = &sm[buf][1][(wc * 32 + nt * 16 + (lane & 15)) * LDP2 + kb];
                bf[nt].u[0] = *(const U4*)(rp + khb);
                bf[nt].u[1] = *(const U4*)(rp + khb + 16);
            }
            #pragma unroll
            for (int mt = 0; mt < 4; mt++)
                #pragma unroll
                for (int nt = 0; nt < 2; nt++)
                    acc[mt][nt] = __builtin_amdgcn_wmma_f32_16x16x32_f16(
                        false, a[mt].v, false, bf[nt].v, (short)0, acc[mt][nt], false, false);
        }
    }

    // epilogue + store: VGPR j -> row (j | 8+j), col = lane&15
    const int cn = lane & 15;
    #pragma unroll
    for (int mt = 0; mt < 4; mt++) {
        #pragma unroll
        for (int nt = 0; nt < 2; nt++) {
            int col = n0 + wc * 32 + nt * 16 + cn;
            if (col >= N) continue;
            int rbase = m0 + wr * 64 + mt * 16 + ((lane < 16) ? 0 : 8);
            #pragma unroll
            for (int j = 0; j < 8; j++) {
                int r = rbase + j;
                float v = acc[mt][nt][j];
                if (EPI == 1) {
                    v += bias[col];
                    v = (v > 20.f) ? v : log1pf(__expf(v));
                } else if (EPI == 2) {
                    v += res[(size_t)r * N + col];
                } else if (EPI == 3) {
                    float t = v + bias[col];
                    float g = 0.5f * t * (1.f + erff(t * 0.70710678118f));
                    v = g + res[(size_t)r * N + col];
                }
                Cout[(size_t)r * N + col] = v;
            }
        }
    }
}

// ---------------------------------------------------------------- chunked scan
// Diagonal linear recurrence => chunk transition is exp(A_n * sum(dt)).
// Pass1: per-chunk end state from h=0 (8x parallel over L). Combine: propagate
// 8 chunk states. Pass2: replay chunk from true start state, emit gated y.
__global__ __launch_bounds__(256) void scan_pass1_kernel(
    const float* __restrict__ dt, const float* __restrict__ xc,
    const float* __restrict__ xdbl, const float* __restrict__ Aneg,
    float* __restrict__ hend, float* __restrict__ sumdt, int dir) {
    const int d = blockIdx.x * 256 + threadIdx.x;
    const int b = blockIdx.y;
    const int chunk = blockIdx.z;

    float An[DSTATE], h[DSTATE];
    #pragma unroll
    for (int n = 0; n < DSTATE; n++) { An[n] = Aneg[(size_t)d * DSTATE + n]; h[n] = 0.f; }
    float sdt = 0.f;

    __shared__ float sB[64][DSTATE];
    for (int sub = 0; sub < CH / 64; sub++) {
        int c0 = chunk * CH + sub * 64;
        __syncthreads();
        for (int e = threadIdx.x; e < 64 * DSTATE; e += 256) {
            int t = e >> 4, k = e & 15;
            int li = dir ? (LL - 1 - (c0 + t)) : (c0 + t);
            sB[t][k] = xdbl[((size_t)(b * LL + li)) * NXP + DTRANK + k];
        }
        __syncthreads();
        for (int t = 0; t < 64; t++) {
            int li = dir ? (LL - 1 - (c0 + t)) : (c0 + t);
            size_t row = (size_t)(b * LL + li);
            float dtv = dt[row * DINNER + d];
            float xv  = xc[row * DINNER + d];
            float dbx = dtv * xv;
            sdt += dtv;
            #pragma unroll
            for (int n = 0; n < DSTATE; n++)
                h[n] = __expf(dtv * An[n]) * h[n] + dbx * sB[t][n];
        }
    }
    size_t base = ((size_t)(b * DINNER + d) * NCH + chunk);
    #pragma unroll
    for (int n = 0; n < DSTATE; n++) hend[base * DSTATE + n] = h[n];
    sumdt[base] = sdt;
}

__global__ __launch_bounds__(256) void scan_combine_kernel(
    const float* __restrict__ Aneg, const float* __restrict__ hend,
    const float* __restrict__ sumdt, float* __restrict__ hstart) {
    const int d = blockIdx.x * 256 + threadIdx.x;
    const int b = blockIdx.y;
    float An[DSTATE], h[DSTATE];
    #pragma unroll
    for (int n = 0; n < DSTATE; n++) { An[n] = Aneg[(size_t)d * DSTATE + n]; h[n] = 0.f; }
    size_t base = (size_t)(b * DINNER + d) * NCH;
    for (int c = 0; c < NCH; c++) {
        #pragma unroll
        for (int n = 0; n < DSTATE; n++) hstart[(base + c) * DSTATE + n] = h[n];
        if (c < NCH - 1) {
            float s = sumdt[base + c];
            #pragma unroll
            for (int n = 0; n < DSTATE; n++)
                h[n] = __expf(An[n] * s) * h[n] + hend[(base + c) * DSTATE + n];
        }
    }
}

__global__ __launch_bounds__(256) void scan_pass2_kernel(
    const float* __restrict__ dt, const float* __restrict__ xc,
    const float* __restrict__ xdbl, const float* __restrict__ xz,
    const float* __restrict__ Aneg, const float* __restrict__ Dv,
    const float* __restrict__ hstart, _Float16* __restrict__ yh, int dir) {
    const int d = blockIdx.x * 256 + threadIdx.x;
    const int b = blockIdx.y;
    const int chunk = blockIdx.z;

    float An[DSTATE], h[DSTATE];
    size_t hbase = ((size_t)(b * DINNER + d) * NCH + chunk) * DSTATE;
    #pragma unroll
    for (int n = 0; n < DSTATE; n++) {
        An[n] = Aneg[(size_t)d * DSTATE + n];
        h[n] = hstart[hbase + n];
    }
    const float Dd = Dv[d];
    const int ycol = dir * DINNER + d;

    __shared__ float sBC[64][2 * DSTATE];
    for (int sub = 0; sub < CH / 64; sub++) {
        int c0 = chunk * CH + sub * 64;
        __syncthreads();
        for (int e = threadIdx.x; e < 64 * 2 * DSTATE; e += 256) {
            int t = e >> 5, k = e & 31;
            int li = dir ? (LL - 1 - (c0 + t)) : (c0 + t);
            sBC[t][k] = xdbl[((size_t)(b * LL + li)) * NXP + DTRANK + k];
        }
        __syncthreads();
        for (int t = 0; t < 64; t++) {
            int li = dir ? (LL - 1 - (c0 + t)) : (c0 + t);
            size_t row = (size_t)(b * LL + li);
            float dtv = dt[row * DINNER + d];
            float xv  = xc[row * DINNER + d];
            float zv  = xz[row * NIN + DINNER + d];
            float dbx = dtv * xv;
            float y = 0.f;
            #pragma unroll
            for (int n = 0; n < DSTATE; n++) {
                h[n] = __expf(dtv * An[n]) * h[n] + dbx * sBC[t][n];
                y += h[n] * sBC[t][DSTATE + n];
            }
            y += xv * Dd;
            y *= zv / (1.f + __expf(-zv));           // * silu(z)
            yh[row * NIN + ycol] = (_Float16)y;
        }
    }
}

// ---------------------------------------------------------------- launcher
extern "C" void kernel_launch(void* const* d_in, const int* in_sizes, int n_in,
                              void* d_out, int out_size, void* d_ws, size_t ws_size,
                              hipStream_t stream) {
    const float* x    = (const float*)d_in[0];
    const float* ln1w = (const float*)d_in[1];
    const float* ln1b = (const float*)d_in[2];
    const float* ln2w = (const float*)d_in[3];
    const float* ln2b = (const float*)d_in[4];
    const float* fcw  = (const float*)d_in[5];
    const float* fcb  = (const float*)d_in[6];
    const float* inpw[2]  = { (const float*)d_in[7],  (const float*)d_in[16] };
    const float* convw[2] = { (const float*)d_in[8],  (const float*)d_in[17] };
    const float* convb[2] = { (const float*)d_in[9],  (const float*)d_in[18] };
    const float* xpw[2]   = { (const float*)d_in[10], (const float*)d_in[19] };
    const float* dtw[2]   = { (const float*)d_in[11], (const float*)d_in[20] };
    const float* dtb[2]   = { (const float*)d_in[12], (const float*)d_in[21] };
    const float* alog[2]  = { (const float*)d_in[13], (const float*)d_in[22] };
    const float* Dvec[2]  = { (const float*)d_in[14], (const float*)d_in[23] };
    const float* outw[2]  = { (const float*)d_in[15], (const float*)d_in[24] };

    size_t off = 0;
    auto carve = [&](size_t bytes) -> void* {
        off = (off + 255) & ~(size_t)255;
        void* p = (char*)d_ws + off;
        off += bytes;
        return p;
    };
    _Float16* xn_h   = (_Float16*)carve((size_t)MM * DMODEL * 2);
    _Float16* inw_h  = (_Float16*)carve((size_t)2 * NIN * DMODEL * 2);
    _Float16* xpw_h  = (_Float16*)carve((size_t)2 * NXP * DINNER * 2);
    _Float16* dtw_h  = (_Float16*)carve((size_t)2 * DINNER * DTPAD * 2);
    _Float16* outw_h = (_Float16*)carve((size_t)DMODEL * NIN * 2);
    _Float16* fcw_h  = (_Float16*)carve((size_t)DMODEL * DMODEL * 2);
    float*    Aneg   = (float*)   carve((size_t)2 * DINNER * DSTATE * 4);
    float*    xzbuf  = (float*)   carve((size_t)2 * MM * NIN * 4);
    float*    xcbuf  = (float*)   carve((size_t)2 * MM * DINNER * 4);
    _Float16* xc_h   = (_Float16*)carve((size_t)2 * MM * DINNER * 2);
    float*    xdbl   = (float*)   carve((size_t)2 * MM * NXP * 4);
    _Float16* dtin_h = (_Float16*)carve((size_t)2 * MM * DTPAD * 2);
    float*    dt32   = (float*)   carve((size_t)2 * MM * DINNER * 4);
    _Float16* y_h    = (_Float16*)carve((size_t)MM * NIN * 2);
    float*    x2     = (float*)   carve((size_t)MM * DMODEL * 4);
    _Float16* x2n_h  = (_Float16*)carve((size_t)MM * DMODEL * 2);
    float*    hend   = (float*)   carve((size_t)BB * DINNER * NCH * DSTATE * 4);
    float*    hstart = (float*)   carve((size_t)BB * DINNER * NCH * DSTATE * 4);
    float*    sumdt  = (float*)   carve((size_t)BB * DINNER * NCH * 4);
    (void)ws_size; (void)n_in; (void)in_sizes; (void)out_size;

    auto cdiv = [](int a, int b) { return (a + b - 1) / b; };

    // weight conversions (f32 -> f16), A = -exp(A_log)
    for (int dir = 0; dir < 2; dir++) {
        cvt_rows_kernel<<<cdiv(NIN * DMODEL, 256), 256, 0, stream>>>(
            inw_h + (size_t)dir * NIN * DMODEL, DMODEL, inpw[dir], NIN, DMODEL);
        cvt_rows_kernel<<<cdiv(NXP * DINNER, 256), 256, 0, stream>>>(
            xpw_h + (size_t)dir * NXP * DINNER, DINNER, xpw[dir], NXP, DINNER);
        cvt_pad_kernel<<<cdiv(DINNER * DTPAD, 256), 256, 0, stream>>>(
            dtw_h + (size_t)dir * DINNER * DTPAD, DTPAD, dtw[dir], DTRANK, DINNER, DTRANK);
        cvt_rows_kernel<<<cdiv(DMODEL * DINNER, 256), 256, 0, stream>>>(
            outw_h + (size_t)dir * DINNER, NIN, outw[dir], DMODEL, DINNER);
        neg_exp_kernel<<<cdiv(DINNER * DSTATE, 256), 256, 0, stream>>>(
            Aneg + (size_t)dir * DINNER * DSTATE, alog[dir], DINNER * DSTATE);
    }
    cvt_rows_kernel<<<cdiv(DMODEL * DMODEL, 256), 256, 0, stream>>>(
        fcw_h, DMODEL, fcw, DMODEL, DMODEL);

    // LN1 -> xn (f16)
    layernorm_f16_kernel<<<MM, 256, 0, stream>>>(x, ln1w, ln1b, xn_h, DMODEL);

    for (int dir = 0; dir < 2; dir++) {
        float*    xz_d  = xzbuf + (size_t)dir * MM * NIN;
        float*    xc_d  = xcbuf + (size_t)dir * MM * DINNER;
        _Float16* xch_d = xc_h  + (size_t)dir * MM * DINNER;
        float*    xd_d  = xdbl  + (size_t)dir * MM * NXP;
        _Float16* dti_d = dtin_h + (size_t)dir * MM * DTPAD;
        float*    dt_d  = dt32  + (size_t)dir * MM * DINNER;
        const float* An_d = Aneg + (size_t)dir * DINNER * DSTATE;

        // in_proj: (2048x768) @ (3072x768)^T
        gemm_wmma_kernel<0><<<dim3(MM / TM, NIN / TN), 256, 0, stream>>>(
            xn_h, inw_h + (size_t)dir * NIN * DMODEL, xz_d, nullptr, nullptr, MM, NIN, DMODEL);
        // depthwise conv + SiLU
        conv_silu_kernel<<<cdiv(MM * DINNER, 256), 256, 0, stream>>>(
            xz_d, convw[dir], convb[dir], xc_d, xch_d, dir);
        // x_proj: (2048x1536) @ (80x1536)^T
        gemm_wmma_kernel<0><<<dim3(MM / TM, cdiv(NXP, TN)), 256, 0, stream>>>(
            xch_d, xpw_h + (size_t)dir * NXP * DINNER, xd_d, nullptr, nullptr, MM, NXP, DINNER);
        // dt slice -> f16, K padded 48->64
        cvt_pad_kernel<<<cdiv(MM * DTPAD, 256), 256, 0, stream>>>(
            dti_d, DTPAD, xd_d, NXP, MM, DTRANK);
        // dt_proj + bias + softplus: (2048x64) @ (1536x64)^T
        gemm_wmma_kernel<1><<<dim3(MM / TM, DINNER / TN), 256, 0, stream>>>(
            dti_d, dtw_h + (size_t)dir * DINNER * DTPAD, dt_d, dtb[dir], nullptr, MM, DINNER, DTPAD);
        // chunked parallel selective scan -> gated y into concat f16 buffer
        scan_pass1_kernel<<<dim3(DINNER / 256, BB, NCH), 256, 0, stream>>>(
            dt_d, xc_d, xd_d, An_d, hend, sumdt, dir);
        scan_combine_kernel<<<dim3(DINNER / 256, BB), 256, 0, stream>>>(
            An_d, hend, sumdt, hstart);
        scan_pass2_kernel<<<dim3(DINNER / 256, BB, NCH), 256, 0, stream>>>(
            dt_d, xc_d, xd_d, xz_d, An_d, Dvec[dir], hstart, y_h, dir);
    }

    // fused out_proj (both dirs, K=3072) + residual x -> x2
    gemm_wmma_kernel<2><<<dim3(MM / TM, DMODEL / TN), 256, 0, stream>>>(
        y_h, outw_h, x2, nullptr, x, MM, DMODEL, NIN);
    // LN2 -> f16
    layernorm_f16_kernel<<<MM, 256, 0, stream>>>(x2, ln2w, ln2b, x2n_h, DMODEL);
    // FC + bias + GELU + residual x2 -> out
    gemm_wmma_kernel<3><<<dim3(MM / TM, DMODEL / TN), 256, 0, stream>>>(
        x2n_h, fcw_h, (float*)d_out, fcb, x2, MM, DMODEL, DMODEL);
}
